// ResidualTask_34617436405990
// MI455X (gfx1250) — matrix operand order repaired and checked by reading, not compiled, not run on hardware
//
#include <hip/hip_runtime.h>

typedef __bf16 v4bf  __attribute__((ext_vector_type(4)));
typedef __bf16 v8bf  __attribute__((ext_vector_type(8)));
typedef __bf16 v16bf __attribute__((ext_vector_type(16)));
typedef float  v8f   __attribute__((ext_vector_type(8)));

#define DDIM 512
#define BROWS 128                 // rows per block
#define BCOLS 256                 // cols per block (N split via blockIdx.y)
#define KC 32                     // K per chunk = WMMA K
#define STR 40                    // LDS row stride (bf16 elems): 20*L mod 64 distinct mult-of-4 -> conflict-free b128
#define NRB (131072 / BROWS)      // 1024 row-blocks

struct bfpair { __bf16 hi; __bf16 lo; };

__device__ __forceinline__ bfpair split_bf16(float v) {
  bfpair p;
  p.hi = (__bf16)v;                    // RNE truncate
  p.lo = (__bf16)(v - (float)p.hi);    // exact residual, re-rounded
  return p;
}

// ---------------------------------------------------------------------------
// Kernel 1: h = x @ W^T + b via bf16x3 split WMMA (fp32 accumulate),
// plus per-block column partial sums of h, h^2 for BatchNorm stats.
// Block: 256 thr = 8 waves; wave w -> rows [bx*128 + 16w, +16) x cols [by*256, +256).
// ---------------------------------------------------------------------------
__global__ __launch_bounds__(256) void gemm_stats(
    const float* __restrict__ x, const float* __restrict__ W,
    const float* __restrict__ bsh, float* __restrict__ h,
    float* __restrict__ pSum, float* __restrict__ pSq) {
  __shared__ __bf16 xh[BROWS * STR];   // 10 KB
  __shared__ __bf16 xlo[BROWS * STR];  // 10 KB
  __shared__ __bf16 wh[BCOLS * STR];   // 20 KB
  __shared__ __bf16 wlo[BCOLS * STR];  // 20 KB
  __shared__ float ssum[BCOLS];        // 1 KB
  __shared__ float ssq[BCOLS];         // 1 KB

  const int tid  = threadIdx.x;
  const int wave = tid >> 5;
  const int lane = tid & 31;
  const int lo   = lane & 15;
  const int hi   = lane >> 4;
  const long row0 = (long)blockIdx.x * BROWS;
  const int  n0   = blockIdx.y * BCOLS;
  const int  m0   = wave * 16;

  v8f acc[16];
#pragma unroll
  for (int j = 0; j < 16; ++j) acc[j] = (v8f)0.0f;

  // staging map: 8 consecutive tids cover one row's 32 K-floats
  const int r8 = tid >> 3;             // 0..31
  const int cc = (tid & 7) * 4;        // 0..28 step 4

  for (int k0 = 0; k0 < DDIM; k0 += KC) {
    __syncthreads();
    // stage x tile: 128 rows x 32 K, split into bf16 hi/lo planes
#pragma unroll
    for (int rr = 0; rr < 4; ++rr) {
      const int r = rr * 32 + r8;
      const float4 v = *(const float4*)(x + (row0 + r) * DDIM + k0 + cc);
      const bfpair p0 = split_bf16(v.x), p1 = split_bf16(v.y);
      const bfpair p2 = split_bf16(v.z), p3 = split_bf16(v.w);
      const v4bf hv = {p0.hi, p1.hi, p2.hi, p3.hi};
      const v4bf lv = {p0.lo, p1.lo, p2.lo, p3.lo};
      *(v4bf*)(&xh [r * STR + cc]) = hv;
      *(v4bf*)(&xlo[r * STR + cc]) = lv;
    }
    // stage W tile: 256 rows (= B cols) x 32 K
#pragma unroll
    for (int rr = 0; rr < 8; ++rr) {
      const int r = rr * 32 + r8;
      const float4 v = *(const float4*)(W + (long)(n0 + r) * DDIM + k0 + cc);
      const bfpair p0 = split_bf16(v.x), p1 = split_bf16(v.y);
      const bfpair p2 = split_bf16(v.z), p3 = split_bf16(v.w);
      const v4bf hv = {p0.hi, p1.hi, p2.hi, p3.hi};
      const v4bf lv = {p0.lo, p1.lo, p2.lo, p3.lo};
      *(v4bf*)(&wh [r * STR + cc]) = hv;
      *(v4bf*)(&wlo[r * STR + cc]) = lv;
    }
    __syncthreads();

    // A frags (16-bit A 16x32 layout: V0-3 = K 8*hi+[0,8), V4-7 = K 16+8*hi+[0,8))
    const __bf16* xrh = &xh [(m0 + lo) * STR + 8 * hi];
    const __bf16* xrl = &xlo[(m0 + lo) * STR + 8 * hi];
    const v8bf ah0 = *(const v8bf*)(xrh);
    const v8bf ah1 = *(const v8bf*)(xrh + 16);
    const v8bf al0 = *(const v8bf*)(xrl);
    const v8bf al1 = *(const v8bf*)(xrl + 16);
    const v16bf ah = __builtin_shufflevector(ah0, ah1, 0,1,2,3,4,5,6,7,8,9,10,11,12,13,14,15);
    const v16bf al = __builtin_shufflevector(al0, al1, 0,1,2,3,4,5,6,7,8,9,10,11,12,13,14,15);

#pragma unroll
    for (int j = 0; j < 16; ++j) {
      // B frag: lane holds K = 8*hi + [0,8) and 16+8*hi + [0,8) of column n = j*16+lo
      const __bf16* brh = &wh [(j * 16 + lo) * STR + 8 * hi];
      const __bf16* brl = &wlo[(j * 16 + lo) * STR + 8 * hi];
      const v8bf bh0 = *(const v8bf*)(brh);
      const v8bf bh1 = *(const v8bf*)(brh + 16);
      const v8bf bl0 = *(const v8bf*)(brl);
      const v8bf bl1 = *(const v8bf*)(brl + 16);
      const v16bf bh = __builtin_shufflevector(bh0, bh1, 0,1,2,3,4,5,6,7,8,9,10,11,12,13,14,15);
      const v16bf bl = __builtin_shufflevector(bl0, bl1, 0,1,2,3,4,5,6,7,8,9,10,11,12,13,14,15);
      acc[j] = __builtin_amdgcn_wmma_f32_16x16x32_bf16(false, ah, false, bh, (short)0, acc[j], false, false);
      acc[j] = __builtin_amdgcn_wmma_f32_16x16x32_bf16(false, al, false, bh, (short)0, acc[j], false, false);
      acc[j] = __builtin_amdgcn_wmma_f32_16x16x32_bf16(false, ah, false, bl, (short)0, acc[j], false, false);
    }
  }

  // ---- epilogue: bias, store h, per-column partial stats ----
  __syncthreads();
  for (int i = tid; i < BCOLS; i += 256) { ssum[i] = 0.0f; ssq[i] = 0.0f; }
  __syncthreads();

#pragma unroll
  for (int j = 0; j < 16; ++j) {
    const int cl  = j * 16 + lo;       // col within block
    const int col = n0 + cl;           // global col
    const float bs = bsh[col];
    float s = 0.0f, q = 0.0f;
#pragma unroll
    for (int i = 0; i < 8; ++i) {
      // C/D layout: VGPR i -> M = i + 8*hi
      const float v = acc[j][i] + bs;
      h[(row0 + m0 + 8 * hi + i) * (long)DDIM + col] = v;
      s += v;
      q += v * v;
    }
    atomicAdd(&ssum[cl], s);   // ds_add_f32
    atomicAdd(&ssq[cl], q);
  }
  __syncthreads();
  for (int i = tid; i < BCOLS; i += 256) {
    pSum[(long)blockIdx.x * DDIM + n0 + i] = ssum[i];
    pSq [(long)blockIdx.x * DDIM + n0 + i] = ssq[i];
  }
}

// ---------------------------------------------------------------------------
// Kernel 2: reduce block partials -> BN affine coefficients a, c
// ---------------------------------------------------------------------------
__global__ __launch_bounds__(256) void bn_finalize(
    const float* __restrict__ pSum, const float* __restrict__ pSq,
    const float* __restrict__ gamma, const float* __restrict__ beta,
    float* __restrict__ acoef, float* __restrict__ ccoef) {
  const int n = blockIdx.x * 256 + threadIdx.x;
  if (n >= DDIM) return;
  float s = 0.0f, q = 0.0f;
#pragma unroll 4
  for (int b = 0; b < NRB; ++b) {
    s += pSum[(long)b * DDIM + n];
    q += pSq [(long)b * DDIM + n];
  }
  const float invB = 1.0f / 131072.0f;
  const float mu  = s * invB;
  const float var = q * invB - mu * mu;
  const float inv = rsqrtf(var + 1e-5f);
  const float a = gamma[n] * inv;
  acoef[n] = a;
  ccoef[n] = beta[n] - mu * a;
}

// ---------------------------------------------------------------------------
// Kernel 3: per-row relu( dot( relu(h*a+c), expert_W[mask] ) + expert_b[mask] )
// ---------------------------------------------------------------------------
__global__ __launch_bounds__(256) void epilogue(
    const float* __restrict__ h, const int* __restrict__ masks,
    const float* __restrict__ acoef, const float* __restrict__ ccoef,
    const float* __restrict__ eW, const float* __restrict__ eb,
    float* __restrict__ out) {
  __shared__ float sa[DDIM], sc[DDIM], sw[8 * DDIM];
  const int tid = threadIdx.x;
  for (int i = tid; i < DDIM; i += 256) { sa[i] = acoef[i]; sc[i] = ccoef[i]; }
  for (int i = tid; i < 8 * DDIM; i += 256) sw[i] = eW[i];
  __syncthreads();

  const int wave = tid >> 5;
  const int lane = tid & 31;
  const long row = (long)blockIdx.x * 8 + wave;
  const int  m   = masks[row];
  const float* hr = h + row * DDIM;
  const float* wr = &sw[m * DDIM];

  float accd = 0.0f;
#pragma unroll
  for (int c4 = 0; c4 < 4; ++c4) {
    const int col = c4 * 128 + lane * 4;
    const float4 hv = *(const float4*)(hr + col);
    accd += fmaxf(hv.x * sa[col + 0] + sc[col + 0], 0.0f) * wr[col + 0];
    accd += fmaxf(hv.y * sa[col + 1] + sc[col + 1], 0.0f) * wr[col + 1];
    accd += fmaxf(hv.z * sa[col + 2] + sc[col + 2], 0.0f) * wr[col + 2];
    accd += fmaxf(hv.w * sa[col + 3] + sc[col + 3], 0.0f) * wr[col + 3];
  }
#pragma unroll
  for (int off = 16; off > 0; off >>= 1) accd += __shfl_xor(accd, off, 32);
  if (lane == 0) out[row] = fmaxf(accd + eb[m], 0.0f);
}

// ---------------------------------------------------------------------------
extern "C" void kernel_launch(void* const* d_in, const int* in_sizes, int n_in,
                              void* d_out, int out_size, void* d_ws, size_t ws_size,
                              hipStream_t stream) {
  const float* x     = (const float*)d_in[0];
  const int*   masks = (const int*)  d_in[1];
  const float* Wsh   = (const float*)d_in[2];
  const float* bsh   = (const float*)d_in[3];
  const float* gamma = (const float*)d_in[4];
  const float* beta  = (const float*)d_in[5];
  const float* eW    = (const float*)d_in[6];
  const float* eb    = (const float*)d_in[7];
  float* out = (float*)d_out;
  float* ws  = (float*)d_ws;

  float* acoef = ws;                                  // 512
  float* ccoef = ws + 512;                            // 512
  float* pSum  = ws + 1024;                           // NRB*512
  float* pSq   = pSum + (long)NRB * DDIM;             // NRB*512
  float* h     = pSq  + (long)NRB * DDIM;             // 131072*512

  dim3 grid(NRB, 2);
  gemm_stats<<<grid, 256, 0, stream>>>(x, Wsh, bsh, h, pSum, pSq);
  bn_finalize<<<2, 256, 0, stream>>>(pSum, pSq, gamma, beta, acoef, ccoef);
  epilogue<<<131072 / 8, 256, 0, stream>>>(h, masks, acoef, ccoef, eW, eb, out);
}
